// BiConvLSTM_26542897889964
// MI455X (gfx1250) — matrix operand order repaired and checked by reading, not compiled, not run on hardware
//
#include <hip/hip_runtime.h>
#include <hip/hip_bf16.h>
#include <cstdint>

typedef __attribute__((ext_vector_type(16))) _Float16 v16h;
typedef __attribute__((ext_vector_type(8)))  _Float16 v8h;
typedef __attribute__((ext_vector_type(8)))  float    v8f;

#define DEV static __device__ __forceinline__

constexpr int Hh = 96, Ww = 96, HW = Hh * Ww;
constexpr int HP = 98, WP = 98, PHW = HP * WP;   // 1-pixel zero halo
constexpr int Bb = 2, Tt = 5, HID = 64, IND = 64;

enum OutMode { OUT_NHWC_F16, OUT_NHWC_F16_RELU, OUT_NHWC_F32, OUT_NCHW_F32, OUT_FINAL_BT };

// ---- WMMA fragment helpers (16x16x32 f16; A and B both use contiguous k = 16*laneHi + s) ----

// A fragment: weights packed row-major [Cout][9*Cin] f16.
// lane<16: row=lane, K = {k0..k0+15}; lane>=16: K = {k0+16..k0+31}.
DEV v16h load_afrag(const _Float16* __restrict__ wrow, int k0, int laneHi) {
  v16h a;
  const _Float16* p = wrow + k0 + laneHi * 16;
  ((v8h*)&a)[0] = *(const v8h*)(p);
  ((v8h*)&a)[1] = *(const v8h*)(p + 8);
  return a;
}

// B fragment: 16 contiguous K halves (laneHi picks K+16 already folded into ptr).
DEV v16h load_bfrag(const _Float16* __restrict__ p) {
  v16h b;
  ((v8h*)&b)[0] = *(const v8h*)(p);
  ((v8h*)&b)[1] = *(const v8h*)(p + 8);
  return b;
}

DEV v8f wmma(v16h a, v16h b, v8f c) {
  return __builtin_amdgcn_wmma_f32_16x16x32_f16(false, a, false, b, (short)0, c,
                                                false, false);
}

// D/C tile store + bias. VGPR r holds rows (r) for lanes 0-15, (r+8) for 16-31.
template <OutMode MODE, int COUTP>
DEV void store_out(void* __restrict__ out, v8f acc, const float* __restrict__ bias,
                   int coutTile, int laneHi, int img, int pin) {
  const int cbase = coutTile * 16 + laneHi * 8;
  v8f bi = *(const v8f*)(bias + cbase);
  if constexpr (MODE == OUT_NHWC_F16 || MODE == OUT_NHWC_F16_RELU) {
    // padded NHWC f16: interior pixel (y,x) lives at (y+1, x+1)
    const int y = pin / Ww, x = pin % Ww;
    v8h ov;
#pragma unroll
    for (int r = 0; r < 8; ++r) {
      float v = acc[r] + bi[r];
      if constexpr (MODE == OUT_NHWC_F16_RELU) v = v > 0.0f ? v : 0.0f;
      ov[r] = (_Float16)v;
    }
    *(v8h*)((_Float16*)out +
            ((size_t)img * PHW + (y + 1) * WP + (x + 1)) * COUTP + cbase) = ov;
  } else if constexpr (MODE == OUT_NHWC_F32) {
    v8f ov;
#pragma unroll
    for (int r = 0; r < 8; ++r) ov[r] = acc[r] + bi[r];
    *(v8f*)((float*)out + ((size_t)img * HW + pin) * COUTP + cbase) = ov;
  } else if constexpr (MODE == OUT_NCHW_F32) {
    float* o = (float*)out + (size_t)img * COUTP * HW + pin;
#pragma unroll
    for (int r = 0; r < 8; ++r) o[(size_t)(cbase + r) * HW] = acc[r] + bi[r];
  } else {  // OUT_FINAL_BT: img = t*Bb + b ; d_out is [B][T][64][H][W]
    const int t = img >> 1, b = img & 1;
    float* o = (float*)out + ((size_t)b * Tt + t) * COUTP * HW + pin;
#pragma unroll
    for (int r = 0; r < 8; ++r) o[(size_t)(cbase + r) * HW] = acc[r] + bi[r];
  }
}

// ---- Implicit-GEMM 3x3 conv (pad=1) over zero-haloed NHWC f16 activations ----
// 2x2 register blocking: each wave owns 2 cout tiles x 2 pixel tiles.
// Halo => branch-free inner loop (EXEC all-ones, as WMMA requires).
template <int CIN, int COUTP, OutMode MODE>
__global__ __launch_bounds__(256) void conv3x3_wmma_k(
    const _Float16* __restrict__ act,   // [NB][HP][WP][CIN], halo = 0
    const _Float16* __restrict__ wk,    // [COUTP][9*CIN], k = tap*CIN + ci
    const float* __restrict__ bias,     // [COUTP]
    void* __restrict__ out, int NB) {
  const int nCoutPairs = COUTP / 32;
  const int nPixPairs  = NB * HW / 32;
  const int wave = (int)((blockIdx.x * blockDim.x + threadIdx.x) >> 5);
  if (wave >= nCoutPairs * nPixPairs) return;  // wave-uniform guard
  const int lane = threadIdx.x & 31;
  const int laneLo = lane & 15, laneHi = lane >> 4;
  const int cp = wave % nCoutPairs;            // neighbor waves share pixels
  const int pp = wave / nCoutPairs;
  const int p0 = pp * 32 + laneLo;             // HW % 32 == 0: same image
  const int img = p0 / HW;
  const int pin0 = p0 % HW, pin1 = pin0 + 16;
  const int y0 = pin0 / Ww, x0 = pin0 % Ww;
  const int y1 = pin1 / Ww, x1 = pin1 % Ww;
  const _Float16* abase = act + (size_t)img * PHW * CIN + laneHi * 16;
  const _Float16* wrow0 = wk + (size_t)(cp * 32 + laneLo) * (9 * CIN);
  const _Float16* wrow1 = wrow0 + (size_t)16 * (9 * CIN);
  v8f a00 = {}, a01 = {}, a10 = {}, a11 = {};
#pragma unroll 3
  for (int tap = 0; tap < 9; ++tap) {
    const int ty = tap / 3, tx = tap % 3;
    const _Float16* s0 = abase + (size_t)((y0 + ty) * WP + (x0 + tx)) * CIN;
    const _Float16* s1 = abase + (size_t)((y1 + ty) * WP + (x1 + tx)) * CIN;
#pragma unroll
    for (int c0 = 0; c0 < CIN; c0 += 32) {
      const v16h af0 = load_afrag(wrow0, tap * CIN + c0, laneHi);
      const v16h af1 = load_afrag(wrow1, tap * CIN + c0, laneHi);
      const v16h bf0 = load_bfrag(s0 + c0);
      const v16h bf1 = load_bfrag(s1 + c0);
      a00 = wmma(af0, bf0, a00);
      a01 = wmma(af0, bf1, a01);
      a10 = wmma(af1, bf0, a10);
      a11 = wmma(af1, bf1, a11);
    }
  }
  store_out<MODE, COUTP>(out, a00, bias, cp * 2,     laneHi, img, pin0);
  store_out<MODE, COUTP>(out, a01, bias, cp * 2,     laneHi, img, pin1);
  store_out<MODE, COUTP>(out, a10, bias, cp * 2 + 1, laneHi, img, pin0);
  store_out<MODE, COUTP>(out, a11, bias, cp * 2 + 1, laneHi, img, pin1);
}

// ---- Plain GEMM over a packed B matrix [NPIX][KTOT] (deformable conv) ----
template <int KTOT, int COUTP, OutMode MODE>
__global__ __launch_bounds__(256) void gemm_wmma_k(
    const _Float16* __restrict__ bmat, const _Float16* __restrict__ wk,
    const float* __restrict__ bias, void* __restrict__ out, int NPIX) {
  const int nCoutPairs = COUTP / 32;
  const int nPixPairs  = NPIX / 32;
  const int wave = (int)((blockIdx.x * blockDim.x + threadIdx.x) >> 5);
  if (wave >= nCoutPairs * nPixPairs) return;
  const int lane = threadIdx.x & 31;
  const int laneLo = lane & 15, laneHi = lane >> 4;
  const int cp = wave % nCoutPairs;
  const int pp = wave / nCoutPairs;
  const int p0 = pp * 32 + laneLo;
  const _Float16* wrow0 = wk + (size_t)(cp * 32 + laneLo) * KTOT;
  const _Float16* wrow1 = wrow0 + (size_t)16 * KTOT;
  const _Float16* brow0 = bmat + (size_t)p0 * KTOT + laneHi * 16;
  const _Float16* brow1 = brow0 + (size_t)16 * KTOT;
  v8f a00 = {}, a01 = {}, a10 = {}, a11 = {};
#pragma unroll 3
  for (int k0 = 0; k0 < KTOT; k0 += 32) {
    const v16h af0 = load_afrag(wrow0, k0, laneHi);
    const v16h af1 = load_afrag(wrow1, k0, laneHi);
    const v16h bf0 = load_bfrag(brow0 + k0);
    const v16h bf1 = load_bfrag(brow1 + k0);
    a00 = wmma(af0, bf0, a00);
    a01 = wmma(af0, bf1, a01);
    a10 = wmma(af1, bf0, a10);
    a11 = wmma(af1, bf1, a11);
  }
  const int img = p0 / HW;
  const int pin0 = p0 % HW, pin1 = pin0 + 16;
  store_out<MODE, COUTP>(out, a00, bias, cp * 2,     laneHi, img, pin0);
  store_out<MODE, COUTP>(out, a01, bias, cp * 2,     laneHi, img, pin1);
  store_out<MODE, COUTP>(out, a10, bias, cp * 2 + 1, laneHi, img, pin0);
  store_out<MODE, COUTP>(out, a11, bias, cp * 2 + 1, laneHi, img, pin1);
}

// ---- Deformable sampling: build packed [pixel][9*64] f16 B-matrix ----
// Thread = (b, pix, g). h lives in xh channels 64..127 (padded NHWC f16).
__global__ __launch_bounds__(256) void dcn_sample(
    const _Float16* __restrict__ xh,   // [B][HP][WP][128]
    const float* __restrict__ om,      // [B][224][H][W] (216 valid)
    _Float16* __restrict__ samp) {     // [B*HW][576], k = tap*64 + g*8 + cg
  const int tid = blockIdx.x * 256 + threadIdx.x;
  if (tid >= Bb * HW * 8) return;
  const int g = tid & 7;
  const int pix = (tid >> 3) % HW;
  const int b = (tid >> 3) / HW;
  const int y = pix / Ww, x = pix % Ww;
  const float* omb = om + (size_t)b * 224 * HW;
  const _Float16* hb = xh + (size_t)b * PHW * 128 + 64 + g * 8;
  _Float16* sp = samp + ((size_t)b * HW + pix) * 576 + g * 8;
  for (int tap = 0; tap < 9; ++tap) {
    const float offy = omb[(size_t)(g * 18 + tap * 2 + 0) * HW + pix];
    const float offx = omb[(size_t)(g * 18 + tap * 2 + 1) * HW + pix];
    float m = omb[(size_t)(144 + g * 9 + tap) * HW + pix];
    m = 1.0f / (1.0f + expf(-m));
    const float py = (float)(y + tap / 3 - 1) + offy;
    const float px = (float)(x + tap % 3 - 1) + offx;
    const float fy = floorf(py), fx = floorf(px);
    const int iy = (int)fy, ix = (int)fx;
    const float wy = py - fy, wx = px - fx;
    float acc[8] = {0.f, 0.f, 0.f, 0.f, 0.f, 0.f, 0.f, 0.f};
    auto corner = [&](int yi, int xi, float w) {
      if (yi >= 0 && yi < Hh && xi >= 0 && xi < Ww) {
        const v8h hv = *(const v8h*)(hb + (size_t)((yi + 1) * WP + (xi + 1)) * 128);
#pragma unroll
        for (int r = 0; r < 8; ++r) acc[r] += w * (float)hv[r];
      }
    };
    corner(iy,     ix,     (1.f - wy) * (1.f - wx));
    corner(iy,     ix + 1, (1.f - wy) * wx);
    corner(iy + 1, ix,     wy * (1.f - wx));
    corner(iy + 1, ix + 1, wy * wx);
    v8h ov;
#pragma unroll
    for (int r = 0; r < 8; ++r) ov[r] = (_Float16)(acc[r] * m);
    *(v8h*)(sp + tap * 64) = ov;
  }
}

// ---- LSTM pointwise: gates packed f32 -> c state, h (into xh + cat buffer) ----
__global__ __launch_bounds__(256) void lstm_pointwise(
    const float* __restrict__ gates,  // [B*HW][256]: ci|cf|co|cg
    float* __restrict__ cst,          // [B*HW][64]
    _Float16* __restrict__ xh,        // [B][HP][WP][128]; h -> channels 64..127
    _Float16* __restrict__ catb,      // [T*B][HP][WP][128]
    int t, int dirOff) {
  const int tid = blockIdx.x * 256 + threadIdx.x;
  if (tid >= Bb * HW * HID) return;
  const int c = tid & 63;
  const int p = tid >> 6;
  const float* gp = gates + (size_t)p * 256;
  const float gi = gp[c], gf = gp[64 + c], go = gp[128 + c], gg = gp[192 + c];
  const float cprev = cst[(size_t)p * 64 + c];
  auto sig = [](float v) { return 1.0f / (1.0f + expf(-v)); };
  const float c2 = sig(gf) * cprev + sig(gi) * tanhf(gg);
  const float h2 = sig(go) * tanhf(c2);
  cst[(size_t)p * 64 + c] = c2;
  const int b = p / HW, pix = p % HW;
  const size_t ppix = (size_t)((pix / Ww) + 1) * WP + (pix % Ww) + 1;
  xh[((size_t)b * PHW + ppix) * 128 + 64 + c] = (_Float16)h2;
  catb[(((size_t)t * Bb + b) * PHW + ppix) * 128 + dirOff + c] = (_Float16)h2;
}

// ---- x ingest: NCHW f32 slice t -> xh channels 0..63 f16 (padded NHWC) ----
__global__ __launch_bounds__(256) void load_x(
    const float* __restrict__ xin, _Float16* __restrict__ xh, int t) {
  const int tid = blockIdx.x * 256 + threadIdx.x;
  if (tid >= Bb * HW * IND) return;
  const int c = tid & 63;
  const int p = tid >> 6;
  const int b = p / HW, pix = p % HW;
  const size_t ppix = (size_t)((pix / Ww) + 1) * WP + (pix % Ww) + 1;
  xh[((size_t)b * PHW + ppix) * 128 + c] =
      (_Float16)xin[(((size_t)b * Tt + t) * IND + c) * HW + pix];
}

// ---- weight packing: OIHW f32 -> [OP][9*I] f16, k = tap*I + ci, zero-pad O..OP ----
__global__ __launch_bounds__(256) void pack_w(
    const float* __restrict__ w, _Float16* __restrict__ wk, int O, int I, int OP) {
  const long tid = blockIdx.x * 256L + threadIdx.x;
  const long K = 9L * I;
  if (tid >= (long)OP * K) return;
  const int o = (int)(tid / K);
  const int k = (int)(tid % K);
  const int tap = k / I, ci = k % I;
  const float v = (o < O) ? w[((size_t)o * I + ci) * 9 + tap] : 0.0f;
  wk[tid] = (_Float16)v;
}

__global__ __launch_bounds__(256) void pad_bias(
    const float* __restrict__ b, float* __restrict__ bp, int O, int OP) {
  const int i = blockIdx.x * 256 + threadIdx.x;
  if (i < OP) bp[i] = (i < O) ? b[i] : 0.0f;
}

__global__ __launch_bounds__(256) void zero_u32(uint32_t* __restrict__ p, long n) {
  const long i = blockIdx.x * 256L + threadIdx.x;
  if (i < n) p[i] = 0u;
}

// ---------------------------------------------------------------------------

extern "C" void kernel_launch(void* const* d_in, const int* in_sizes, int n_in,
                              void* d_out, int out_size, void* d_ws, size_t ws_size,
                              hipStream_t stream) {
  (void)in_sizes; (void)n_in; (void)out_size; (void)ws_size;
  const float* x_in   = (const float*)d_in[0];
  const float* fuse_w = (const float*)d_in[1];
  const float* fuse_b = (const float*)d_in[2];
  const float* om_w   = (const float*)d_in[3];
  const float* om_b   = (const float*)d_in[4];
  const float* dcn_w  = (const float*)d_in[5];
  const float* dcn_b  = (const float*)d_in[6];
  const float* conv_w = (const float*)d_in[7];
  const float* conv_b = (const float*)d_in[8];
  const float* cat_w  = (const float*)d_in[9];
  const float* cat_b  = (const float*)d_in[10];

  size_t off = 0;
  auto alloc = [&](size_t bytes) -> void* {
    off = (off + 255) & ~(size_t)255;
    void* p = (char*)d_ws + off;
    off += bytes;
    return p;
  };
  _Float16* wk_fuse = (_Float16*)alloc((size_t)64 * 1152 * 2);
  _Float16* wk_om   = (_Float16*)alloc((size_t)224 * 576 * 2);
  _Float16* wk_dcn  = (_Float16*)alloc((size_t)128 * 576 * 2);
  _Float16* wk_conv = (_Float16*)alloc((size_t)256 * 1152 * 2);
  _Float16* wk_cat  = (_Float16*)alloc((size_t)64 * 1152 * 2);
  float*    omb_pad = (float*)alloc((size_t)224 * 4);
  _Float16* xh      = (_Float16*)alloc((size_t)Bb * PHW * 128 * 2);
  _Float16* comb    = (_Float16*)alloc((size_t)Bb * PHW * 64 * 2);
  float*    om      = (float*)alloc((size_t)Bb * 224 * HW * 4);
  _Float16* samp    = (_Float16*)alloc((size_t)Bb * HW * 576 * 2);
  _Float16* fused   = (_Float16*)alloc((size_t)Bb * PHW * 128 * 2);
  float*    gates   = (float*)alloc((size_t)Bb * HW * 256 * 4);
  float*    cst     = (float*)alloc((size_t)Bb * HW * 64 * 4);
  _Float16* catb    = (_Float16*)alloc((size_t)Tt * Bb * PHW * 128 * 2);

  auto blocks1 = [](long n) { return dim3((unsigned)((n + 255) / 256)); };
  auto gemmGrid = [](int coutp, long npix) {
    const long waves = (long)(coutp / 32) * (npix / 32);
    return dim3((unsigned)((waves * 32 + 255) / 256));
  };

  // Weight packing (f32 OIHW -> f16 implicit-GEMM rows), every call (stateless).
  pack_w<<<blocks1(64L * 1152), 256, 0, stream>>>(fuse_w, wk_fuse, 64, 128, 64);
  pack_w<<<blocks1(224L * 576), 256, 0, stream>>>(om_w, wk_om, 216, 64, 224);
  pack_w<<<blocks1(128L * 576), 256, 0, stream>>>(dcn_w, wk_dcn, 128, 64, 128);
  pack_w<<<blocks1(256L * 1152), 256, 0, stream>>>(conv_w, wk_conv, 256, 128, 256);
  pack_w<<<blocks1(64L * 1152), 256, 0, stream>>>(cat_w, wk_cat, 64, 128, 64);
  pad_bias<<<blocks1(224), 256, 0, stream>>>(om_b, omb_pad, 216, 224);

  // Zero haloed activation buffers once per launch; interior is re-written each
  // step, halo stays zero across graph replays of this launch's work.
  zero_u32<<<blocks1((long)Bb * PHW * 32), 256, 0, stream>>>(
      (uint32_t*)comb, (long)Bb * PHW * 32);
  zero_u32<<<blocks1((long)Bb * PHW * 64), 256, 0, stream>>>(
      (uint32_t*)fused, (long)Bb * PHW * 64);
  zero_u32<<<blocks1((long)Tt * Bb * PHW * 64), 256, 0, stream>>>(
      (uint32_t*)catb, (long)Tt * Bb * PHW * 64);

  for (int dir = 0; dir < 2; ++dir) {
    zero_u32<<<blocks1((long)Bb * PHW * 64), 256, 0, stream>>>(
        (uint32_t*)xh, (long)Bb * PHW * 64);  // h := 0 (and halo := 0)
    zero_u32<<<blocks1((long)Bb * HW * 64), 256, 0, stream>>>(
        (uint32_t*)cst, (long)Bb * HW * 64);  // c := 0
    for (int s = 0; s < Tt; ++s) {
      const int t = dir ? (Tt - 1 - s) : s;
      load_x<<<blocks1((long)Bb * HW * IND), 256, 0, stream>>>(x_in, xh, t);
      // fuse: concat[x,h](128) -> combined(64), padded NHWC f16
      conv3x3_wmma_k<128, 64, OUT_NHWC_F16>
          <<<gemmGrid(64, (long)Bb * HW), 256, 0, stream>>>(xh, wk_fuse, fuse_b, comb, Bb);
      // offsets/mask: combined(64) -> om(216 pad 224), NCHW f32
      conv3x3_wmma_k<64, 224, OUT_NCHW_F32>
          <<<gemmGrid(224, (long)Bb * HW), 256, 0, stream>>>(comb, wk_om, omb_pad, om, Bb);
      // deformable bilinear sampling of h -> packed B matrix
      dcn_sample<<<blocks1((long)Bb * HW * 8), 256, 0, stream>>>(xh, om, samp);
      // deformable conv GEMM: samp(576) -> fused(128), relu, padded NHWC f16
      gemm_wmma_k<576, 128, OUT_NHWC_F16_RELU>
          <<<gemmGrid(128, (long)Bb * HW), 256, 0, stream>>>(samp, wk_dcn, dcn_b, fused,
                                                             Bb * HW);
      // gates: fused(128) -> cc(256), packed NHWC f32
      conv3x3_wmma_k<128, 256, OUT_NHWC_F32>
          <<<gemmGrid(256, (long)Bb * HW), 256, 0, stream>>>(fused, wk_conv, conv_b, gates, Bb);
      // pointwise LSTM; h feeds back into xh and into the cat buffer
      lstm_pointwise<<<blocks1((long)Bb * HW * HID), 256, 0, stream>>>(
          gates, cst, xh, catb, t, dir ? 64 : 0);
    }
  }
  // final conv over concat[fwd,bwd](128) for all T*B images -> d_out [B][T][64][H][W]
  conv3x3_wmma_k<128, 64, OUT_FINAL_BT>
      <<<gemmGrid(64, (long)Tt * Bb * HW), 256, 0, stream>>>(catb, wk_cat, cat_b, d_out,
                                                             Tt * Bb);
}
